// GATNet_16853451669884
// MI455X (gfx1250) — matrix operand order repaired and checked by reading, not compiled, not run on hardware
//
#include <hip/hip_runtime.h>
#include <hip/hip_bf16.h>

typedef __bf16 bf16_t;
typedef __attribute__((ext_vector_type(8)))  __bf16 v8bf;
typedef __attribute__((ext_vector_type(16))) __bf16 v16bf;
typedef __attribute__((ext_vector_type(8)))  float  v8f;

#define GAT_N     4096
#define GAT_FIN   1386
#define GAT_KP    1408      // FIN padded to multiple of 32 (rows stay 16B aligned)
#define GAT_NHID  1024
#define GAT_NH    8
#define GAT_HC    8192
#define GAT_CODE  64
#define LRELU_A   0.2f

// ---------------- epilogue modes ----------------
#define EPI_MASK      0   // uint8 mask = (acc >= 0.8)
#define EPI_BF16T     1   // store bf16 transposed: out[n*ldc + m]
#define EPI_ELU_BF16  2   // store elu(acc) bf16 row-major
#define EPI_TANH_F32  3   // store tanh(acc) fp32 row-major

// GEMM tiling: block 64(M) x 256(N), K-step 32. 8 waves in 2x4 grid; wave tile
// 32x64 = 2x4 WMMA 16x16x32 bf16 fragments -> 8 v_wmma per wave per K-step.
// 3-stage LDS pipeline fed by async global->LDS copies (5 per thread per tile).
#define BM 64
#define BN 256
#define LDSR 40                      // padded row stride (elems): 32 data + 8 pad
#define STAGE_ELEMS ((BM + BN) * LDSR)
#define STAGE_BYTES (STAGE_ELEMS * 2)

// CDNA5 async global->LDS copy (tracked by ASYNCcnt, no VGPR staging)
__device__ __forceinline__ void async_copy16(unsigned lds_addr, const void* gptr) {
    asm volatile("global_load_async_to_lds_b128 %0, %1, off"
                 :: "v"(lds_addr), "v"(gptr) : "memory");
}
// Async loads complete in order: <=5 outstanding means every batch but the
// newest (5 instructions/tile) has landed.
__device__ __forceinline__ void wait_async_prev() {
    asm volatile("s_wait_asynccnt 0x5" ::: "memory");
}
__device__ __forceinline__ void wait_async_all() {
    asm volatile("s_wait_asynccnt 0x0" ::: "memory");
}

template <int EPI>
__global__ __launch_bounds__(256)
void gemm_wmma(const bf16_t* __restrict__ A,   // M x K, row-major (lda = K, K%32==0, K>=64)
               const bf16_t* __restrict__ Bcm, // column n at Bcm + n*K; padded to >= gridDim.x*256 cols
               int M, int Ncols, int K,
               void* __restrict__ Cout, int ldc)
{
    __shared__ bf16_t smem[3 * STAGE_ELEMS];   // triple-buffered A(64x32) + B(256x32) tiles

    const int tid     = threadIdx.x;
    const int lane    = tid & 31;
    const int wid     = tid >> 5;
    const int waveRow = wid >> 2;   // 0..1  (rows of 32)
    const int waveCol = wid & 3;    // 0..3  (cols of 64)
    const int l16     = lane & 15;
    const int halfSel = (lane < 16) ? 0 : 1;

    const int blockM = blockIdx.y * BM;
    const int blockN = blockIdx.x * BN;

    // per-thread async-copy source pointers / LDS byte destinations
    const unsigned smem_base = (unsigned)(uintptr_t)(const void*)smem;  // low 32 bits = LDS offset
    const bf16_t* aPtr = A   + (size_t)(blockM + (tid >> 2)) * K + (tid & 3) * 8;  // 1x16B chunk
    const bf16_t* bPtr = Bcm + (size_t)(blockN + tid) * K;                          // 4x16B chunks
    const unsigned aDst = smem_base + ((unsigned)(tid >> 2) * LDSR + (tid & 3) * 8) * 2;
    const unsigned bDst = smem_base + (BM * LDSR + (unsigned)tid * LDSR) * 2;

    v8f acc[2][4];
#pragma unroll
    for (int i = 0; i < 2; ++i)
#pragma unroll
        for (int j = 0; j < 4; ++j)
#pragma unroll
            for (int v = 0; v < 8; ++v) acc[i][j][v] = 0.0f;

    auto issue = [&](int stg, int k0) {     // 5 async b128 copies = one tile batch
        const unsigned so = (unsigned)stg * STAGE_BYTES;
        async_copy16(aDst + so, aPtr + k0);
        const bf16_t* bp = bPtr + k0;
        const unsigned bd = bDst + so;
        async_copy16(bd,      bp);
        async_copy16(bd + 16, bp + 8);
        async_copy16(bd + 32, bp + 16);
        async_copy16(bd + 48, bp + 24);
    };

    // prime two tiles (all our K >= 64)
    issue(0, 0);
    issue(1, 32);

    int stage = 0;
    for (int k0 = 0; k0 < K; k0 += 32) {
        // oldest in-flight batch (this stage) must have landed; keep newest flying
        if (k0 + 32 < K) wait_async_prev();
        else             wait_async_all();
        __syncthreads();   // all waves' copies into this stage visible; previous stage free

        const bf16_t* As = smem + stage * STAGE_ELEMS;
        const bf16_t* Bs = As + BM * LDSR;

        // fragments per documented CDNA5 WMMA VGPR layouts
        v16bf afrag[2], bfrag[4];
#pragma unroll
        for (int fm = 0; fm < 2; ++fm) {
            const int mrow = waveRow * 32 + fm * 16 + l16;
            const int kb   = halfSel * 8;      // lanes0-15: K{0-7,16-23}; lanes16-31: K{8-15,24-31}
            v8bf lo = *(const v8bf*)&As[mrow * LDSR + kb];
            v8bf hi = *(const v8bf*)&As[mrow * LDSR + kb + 16];
            afrag[fm] = __builtin_shufflevector(lo, hi, 0,1,2,3,4,5,6,7,8,9,10,11,12,13,14,15);
        }
#pragma unroll
        for (int fn = 0; fn < 4; ++fn) {
            const int ncol = waveCol * 64 + fn * 16 + l16;
            const int kb   = halfSel * 8;      // lanes0-15: K{0-7,16-23}; lanes16-31: K{8-15,24-31}
            v8bf lo = *(const v8bf*)&Bs[ncol * LDSR + kb];
            v8bf hi = *(const v8bf*)&Bs[ncol * LDSR + kb + 16];
            bfrag[fn] = __builtin_shufflevector(lo, hi, 0,1,2,3,4,5,6,7,8,9,10,11,12,13,14,15);
        }
#pragma unroll
        for (int fm = 0; fm < 2; ++fm)
#pragma unroll
            for (int fn = 0; fn < 4; ++fn)
                acc[fm][fn] = __builtin_amdgcn_wmma_f32_16x16x32_bf16(
                    false, afrag[fm], false, bfrag[fn], (short)0, acc[fm][fn], false, false);

        // prefetch tile t+2 into the stage that just became free
        if (k0 + 64 < K) {
            int s2 = stage + 2; if (s2 >= 3) s2 -= 3;
            issue(s2, k0 + 64);
        }
        stage = (stage == 2) ? 0 : stage + 1;
    }

    // ---- epilogue: lane l16 = column, VGPR v = row (+8 for lanes 16-31) ----
#pragma unroll
    for (int fm = 0; fm < 2; ++fm)
#pragma unroll
        for (int fn = 0; fn < 4; ++fn)
#pragma unroll
            for (int v = 0; v < 8; ++v) {
                const int m = blockM + waveRow * 32 + fm * 16 + halfSel * 8 + v;
                const int n = blockN + waveCol * 64 + fn * 16 + l16;
                if (m < M && n < Ncols) {
                    const float a = acc[fm][fn][v];
                    if (EPI == EPI_MASK) {
                        ((unsigned char*)Cout)[(size_t)m * ldc + n] = (a >= 0.8f) ? 1 : 0;
                    } else if (EPI == EPI_BF16T) {
                        ((bf16_t*)Cout)[(size_t)n * ldc + m] = (bf16_t)a;
                    } else if (EPI == EPI_ELU_BF16) {
                        const float e = a > 0.0f ? a : (__expf(a) - 1.0f);
                        ((bf16_t*)Cout)[(size_t)m * ldc + n] = (bf16_t)e;
                    } else { // EPI_TANH_F32
                        ((float*)Cout)[(size_t)m * ldc + n] = tanhf(a);
                    }
                }
            }
}

// ---- row-normalize x; emit bf16 x (padded) and bf16 xn (padded) ----
__global__ __launch_bounds__(256)
void convert_x(const float* __restrict__ x, bf16_t* __restrict__ xh,
               bf16_t* __restrict__ xnh, int Fin, int KP)
{
    const int m   = blockIdx.x;
    const int tid = threadIdx.x;
    __shared__ float red[256];
    float ss = 0.0f;
    for (int k = tid; k < Fin; k += 256) {
        const float v = x[(size_t)m * Fin + k];
        ss += v * v;
    }
    red[tid] = ss;
    __syncthreads();
    for (int s = 128; s > 0; s >>= 1) {
        if (tid < s) red[tid] += red[tid + s];
        __syncthreads();
    }
    const float inv = 1.0f / fmaxf(sqrtf(red[0]), 1e-12f);
    for (int k = tid; k < KP; k += 256) {
        const float v = (k < Fin) ? x[(size_t)m * Fin + k] : 0.0f;
        xh [(size_t)m * KP + k] = (bf16_t)v;
        xnh[(size_t)m * KP + k] = (bf16_t)(v * inv);
    }
}

// ---- LDS-tiled transpose+convert: W (K x Ncols fp32 row-major) -> Wcm (Ncols x KP bf16 col-major) ----
__global__ __launch_bounds__(256)
void transpose_cvt(const float* __restrict__ W, bf16_t* __restrict__ Wcm,
                   int K, int Ncols, int KP)
{
    __shared__ float t[32][33];
    const int kb = blockIdx.y * 32, nb = blockIdx.x * 32;
    const int tx = threadIdx.x & 31, ty = threadIdx.x >> 5;   // ty: 0..7
#pragma unroll
    for (int i = 0; i < 4; ++i) {
        const int k = kb + ty * 4 + i;
        const int n = nb + tx;
        t[ty * 4 + i][tx] = (k < K && n < Ncols) ? W[(size_t)k * Ncols + n] : 0.0f;
    }
    __syncthreads();
#pragma unroll
    for (int i = 0; i < 4; ++i) {
        const int n = nb + ty * 4 + i;
        const int k = kb + tx;
        if (n < Ncols && k < KP) Wcm[(size_t)n * KP + k] = (bf16_t)t[tx][ty * 4 + i];
    }
}

// ---- f1[m] = sum_k Wh[m,k]*a[k]; f2[m] = sum_k Wh[m,k]*a[H+k]; Wh stored column-major ----
__global__ __launch_bounds__(256)
void rowdot2(const bf16_t* __restrict__ Whcm, const float* __restrict__ a,
             int H, int M, float* __restrict__ f1, float* __restrict__ f2)
{
    const int m = blockIdx.x * 256 + threadIdx.x;
    if (m >= M) return;
    float s1 = 0.0f, s2 = 0.0f;
    for (int k = 0; k < H; ++k) {
        const float w = (float)Whcm[(size_t)k * M + m];  // coalesced over m
        s1 += w * a[k];
        s2 += w * a[H + k];
    }
    f1[m] = s1;
    f2[m] = s2;
}

// ---- fused e = lrelu(f1[m]+f2[j]) -> mask -> row softmax -> att (bf16) ----
__global__ __launch_bounds__(256)
void att_softmax(const float* __restrict__ f1, const float* __restrict__ f2,
                 const unsigned char* __restrict__ mask, bf16_t* __restrict__ att, int N)
{
    const int m   = blockIdx.x;
    const int tid = threadIdx.x;
    const int per = N >> 8;
    const float fm1 = f1[m];
    float ev[16];
    float mx = -3.0e38f;
    for (int i = 0; i < per; ++i) {
        const int j = tid + i * 256;
        float e = -3.0e38f;
        if (mask[(size_t)m * N + j]) {
            const float s = fm1 + f2[j];
            e = (s > 0.0f) ? s : LRELU_A * s;
        }
        ev[i] = e;
        mx = fmaxf(mx, e);
    }
    __shared__ float red[256];
    red[tid] = mx;
    __syncthreads();
    for (int s = 128; s > 0; s >>= 1) {
        if (tid < s) red[tid] = fmaxf(red[tid], red[tid + s]);
        __syncthreads();
    }
    const float rowmax = red[0];
    __syncthreads();
    float sum = 0.0f;
    for (int i = 0; i < per; ++i) {
        const float e = (ev[i] <= -1.0e38f) ? 0.0f : __expf(ev[i] - rowmax);
        ev[i] = e;
        sum += e;
    }
    red[tid] = sum;
    __syncthreads();
    for (int s = 128; s > 0; s >>= 1) {
        if (tid < s) red[tid] += red[tid + s];
        __syncthreads();
    }
    const float inv = 1.0f / red[0];
    for (int i = 0; i < per; ++i)
        att[(size_t)m * N + tid + i * 256] = (bf16_t)(ev[i] * inv);
}

extern "C" void kernel_launch(void* const* d_in, const int* in_sizes, int n_in,
                              void* d_out, int out_size, void* d_ws, size_t ws_size,
                              hipStream_t stream)
{
    (void)in_sizes; (void)n_in; (void)out_size; (void)ws_size;
    const float* x     = (const float*)d_in[0];   // N x FIN
    const float* W_att = (const float*)d_in[1];   // 8 x FIN x NHID
    const float* a_att = (const float*)d_in[2];   // 8 x 2*NHID
    const float* W_out = (const float*)d_in[3];   // HC x CODE
    const float* a_out = (const float*)d_in[4];   // 2*CODE
    float* out = (float*)d_out;                   // N x CODE fp32

    const size_t N = GAT_N, KP = GAT_KP, NHID = GAT_NHID, HC = GAT_HC;
    const size_t CODEP = 256;                     // CODE padded to BN columns (unguarded B loads)

    // ---- workspace layout (256B aligned slabs) ----
    char* ws = (char*)d_ws;
    size_t off = 0;
    auto alloc = [&](size_t bytes) -> char* {
        char* p = ws + off;
        off = (off + bytes + 255) & ~(size_t)255;
        return p;
    };
    bf16_t*        xh     = (bf16_t*)alloc(N * KP * 2);
    bf16_t*        xnh    = (bf16_t*)alloc(N * KP * 2);
    unsigned char* mask   = (unsigned char*)alloc(N * N);
    bf16_t*        Wcm    = (bf16_t*)alloc(NHID * KP * 2);
    bf16_t*        Whcm   = (bf16_t*)alloc(NHID * N * 2);
    float*         f1     = (float*)alloc(N * 4);
    float*         f2     = (float*)alloc(N * 4);
    bf16_t*        att    = (bf16_t*)alloc(N * N * 2);
    bf16_t*        hcat   = (bf16_t*)alloc(N * HC * 2);
    bf16_t*        Woutcm = (bf16_t*)alloc(CODEP * HC * 2);    // padded to 256 cols
    bf16_t*        WhOcm  = (bf16_t*)alloc(CODEP * N * 2);     // padded to 256 cols

    // 1) normalize + convert
    convert_x<<<GAT_N, 256, 0, stream>>>(x, xh, xnh, GAT_FIN, GAT_KP);

    // 2) sim = xn @ xn^T, threshold 0.8 -> mask  (col-major B == xn row-major)
    gemm_wmma<EPI_MASK><<<dim3(GAT_N / 256, GAT_N / 64), 256, 0, stream>>>(
        xnh, xnh, GAT_N, GAT_N, GAT_KP, (void*)mask, GAT_N);

    // 3) attention heads
    for (int h = 0; h < GAT_NH; ++h) {
        transpose_cvt<<<dim3(GAT_NHID / 32, GAT_KP / 32), 256, 0, stream>>>(
            W_att + (size_t)h * GAT_FIN * GAT_NHID, Wcm, GAT_FIN, GAT_NHID, GAT_KP);

        gemm_wmma<EPI_BF16T><<<dim3(GAT_NHID / 256, GAT_N / 64), 256, 0, stream>>>(
            xh, Wcm, GAT_N, GAT_NHID, GAT_KP, (void*)Whcm, GAT_N);

        rowdot2<<<GAT_N / 256, 256, 0, stream>>>(
            Whcm, a_att + (size_t)h * 2 * GAT_NHID, GAT_NHID, GAT_N, f1, f2);

        att_softmax<<<GAT_N, 256, 0, stream>>>(f1, f2, mask, att, GAT_N);

        gemm_wmma<EPI_ELU_BF16><<<dim3(GAT_NHID / 256, GAT_N / 64), 256, 0, stream>>>(
            att, Whcm, GAT_N, GAT_NHID, GAT_N, (void*)(hcat + (size_t)h * GAT_NHID), GAT_HC);
    }

    // 4) output head
    transpose_cvt<<<dim3((GAT_CODE + 31) / 32, GAT_HC / 32), 256, 0, stream>>>(
        W_out, Woutcm, GAT_HC, GAT_CODE, GAT_HC);

    gemm_wmma<EPI_BF16T><<<dim3(1, GAT_N / 64), 256, 0, stream>>>(
        hcat, Woutcm, GAT_N, GAT_CODE, GAT_HC, (void*)WhOcm, GAT_N);

    rowdot2<<<GAT_N / 256, 256, 0, stream>>>(WhOcm, a_out, GAT_CODE, GAT_N, f1, f2);

    att_softmax<<<GAT_N, 256, 0, stream>>>(f1, f2, mask, att, GAT_N);

    gemm_wmma<EPI_TANH_F32><<<dim3(1, GAT_N / 64), 256, 0, stream>>>(
        att, WhOcm, GAT_N, GAT_CODE, GAT_N, (void*)out, GAT_CODE);
}
